// GraphAttention_47579647705278
// MI455X (gfx1250) — matrix operand order repaired and checked by reading, not compiled, not run on hardware
//
#include <hip/hip_runtime.h>

// ---------------- constants ----------------
#define Lq   1024
#define Bc   8
#define Ec   512
#define Hc   8
#define Dh   64

typedef __attribute__((ext_vector_type(16))) __bf16 v16bf;
typedef __attribute__((ext_vector_type(8)))  float  v8f;

union Frag16 { v16bf v; unsigned d[8]; uint4 q[2]; };

// fp32 -> bf16 (round to nearest even)
__device__ __forceinline__ unsigned short f2bf(float f) {
    unsigned u = __float_as_uint(f);
    unsigned r = u + 0x7FFFu + ((u >> 16) & 1u);
    return (unsigned short)(r >> 16);
}
__device__ __forceinline__ unsigned pack2(float lo, float hi) {
    return (unsigned)f2bf(lo) | ((unsigned)f2bf(hi) << 16);
}

// ============================================================
// Kernel 1: qkv = x @ in_proj_w.T + b ;  scatter to Q,K,V(bf16)
//   x: [L*B, 512] fp32, W: [1536, 512] fp32 (row-major)
//   Q: [B,H,L,D] bf16 (pre-scaled 1/sqrt(D))
//   K: [B,H,L,D] bf16
//   V: [B,H,D,L] bf16  (TRANSPOSED so attention B-frags are b128 loads)
// One wave per 16x16 output tile, K-loop step 32.
// ============================================================
__global__ __launch_bounds__(256)
void qkv_kernel(const float* __restrict__ x, const float* __restrict__ W,
                const float* __restrict__ bias,
                unsigned short* __restrict__ Qb,
                unsigned short* __restrict__ Kb,
                unsigned short* __restrict__ Vt)
{
    const int lane = threadIdx.x & 31;
    const int wid  = blockIdx.x * (blockDim.x >> 5) + (threadIdx.x >> 5);
    const int TN = (3 * Ec) / 16;            // 96 col tiles
    const int tile_m = wid / TN, tile_n = wid % TN;
    if (tile_m >= (Lq * Bc) / 16) return;
    const int m0 = tile_m * 16, n0 = tile_n * 16;
    const int half = lane >> 4, l15 = lane & 15;

    v8f acc = {};
    const float* arow = x + (size_t)(m0 + l15) * Ec;      // A row (lane = M)
    const float* brow = W + (size_t)(n0 + l15) * Ec;      // B col n -> W row
    for (int k0 = 0; k0 < Ec; k0 += 32) {
        Frag16 a, b;
#pragma unroll
        for (int v = 0; v < 8; ++v) {
            const int ka = k0 + half * 16 + 2 * v;
            a.d[v] = pack2(arow[ka], arow[ka + 1]);
            const int kb = k0 + half * 16 + 2 * v;
            b.d[v] = pack2(brow[kb], brow[kb + 1]);
        }
        acc = __builtin_amdgcn_wmma_f32_16x16x32_bf16(false, a.v, false, b.v,
                                                      (short)0, acc, false, false);
    }

    // column-derived values are uniform in r (and seg is uniform per wave)
    const int c   = n0 + l15;                // column in [0,1536)
    const int seg = c >> 9;                  // 0=Q 1=K 2=V
    const int e   = c & (Ec - 1);
    const int h   = e >> 6, d = e & 63;
    const float scale = (seg == 0) ? 0.125f : 1.0f;       // 1/sqrt(64) on Q
    const float bv = bias[c];
#pragma unroll
    for (int r = 0; r < 8; ++r) {
        const int m   = r + 8 * half;
        const int row = m0 + m;              // row = l*B + b
        const int l   = row >> 3, bb = row & 7;
        const unsigned short val = f2bf((acc[r] + bv) * scale);
        if (seg == 2)       // V transposed: [B,H,D,L]
            Vt[(((size_t)(bb * Hc + h)) * Dh + d) * Lq + l] = val;
        else {
            unsigned short* dst = (seg == 0) ? Qb : Kb;
            dst[(((size_t)(bb * Hc + h)) * Lq + l) * Dh + d] = val;
        }
    }
}

// ============================================================
// Kernel 2: fused flash-attention with edge bias.
// One wave (32 thr) per (b, h, 16-query block). Key blocks of 32.
// All bf16 fragments loaded as pairs of b128.
// ============================================================
__global__ __launch_bounds__(32)
void attn_kernel(const unsigned short* __restrict__ Qb,
                 const unsigned short* __restrict__ Kb,
                 const unsigned short* __restrict__ Vt,   // [B,H,D,L]
                 const int* __restrict__ edge,            // [B,L,L] int32
                 const float* __restrict__ attn_mask,     // [L,L]
                 const unsigned char* __restrict__ kpm,   // [B,L] bool
                 const float* __restrict__ edge_table,    // [V,H]
                 unsigned short* __restrict__ AO)         // [L*B, 512] bf16
{
    __shared__ float etab[16];
    __shared__ float Pl[16][33];

    const int lane = threadIdx.x & 31;
    const int wg   = blockIdx.x;
    const int qblk = wg & (Lq / 16 - 1);       // 64 q-blocks
    const int bh   = wg >> 6;                  // [0, 64)
    const int b    = bh >> 3, h = bh & 7;
    const int q0   = qblk * 16;
    const int half = lane >> 4, l15 = lane & 15;
    const float NEG_INF = -__builtin_inff();

    if (lane < 16) etab[lane] = edge_table[lane * Hc + h];
    __syncthreads();

    // Q A-fragments (16 rows x 64 d, two 16x32 frags, b128 loads)
    // A layout: k = half*16 + 2v + p
    //   -> q[0] covers k = half*16 .. half*16+7 ; q[1] covers half*16+8 ..
    Frag16 aq0, aq1;
    {
        const unsigned short* qrow = Qb + ((size_t)bh * Lq + (q0 + l15)) * Dh;
        aq0.q[0] = *(const uint4*)(qrow + half * 16);
        aq0.q[1] = *(const uint4*)(qrow + 8 + half * 16);
        aq1.q[0] = *(const uint4*)(qrow + 32 + half * 16);
        aq1.q[1] = *(const uint4*)(qrow + 40 + half * 16);
    }

    float mrow[8], lrow[8];
    v8f o[4] = {};
#pragma unroll
    for (int r = 0; r < 8; ++r) { mrow[r] = NEG_INF; lrow[r] = 0.f; }

    for (int j0 = 0; j0 < Lq; j0 += 32) {
        // ---- scores: S = Q * K^T for 32 keys (two 16x16 C tiles)
        // B layout: lane = key, k(d) = half*16 + 2v + p  -> 16 contiguous d
        v8f s0 = {}, s1 = {};
        {
            const unsigned short* k0p = Kb + ((size_t)bh * Lq + (j0 + l15)) * Dh + half * 16;
            const unsigned short* k1p = k0p + 16 * Dh;
            if (j0 + 32 < Lq) {                 // prefetch next key block
                __builtin_prefetch(k0p + 32 * Dh, 0, 0);
                __builtin_prefetch(k1p + 32 * Dh, 0, 0);
            }
            Frag16 bk;
            bk.q[0] = *(const uint4*)(k0p);
            bk.q[1] = *(const uint4*)(k0p + 8);
            s0 = __builtin_amdgcn_wmma_f32_16x16x32_bf16(false, aq0.v, false, bk.v, (short)0, s0, false, false);
            bk.q[0] = *(const uint4*)(k0p + 32);
            bk.q[1] = *(const uint4*)(k0p + 40);
            s0 = __builtin_amdgcn_wmma_f32_16x16x32_bf16(false, aq1.v, false, bk.v, (short)0, s0, false, false);
            bk.q[0] = *(const uint4*)(k1p);
            bk.q[1] = *(const uint4*)(k1p + 8);
            s1 = __builtin_amdgcn_wmma_f32_16x16x32_bf16(false, aq0.v, false, bk.v, (short)0, s1, false, false);
            bk.q[0] = *(const uint4*)(k1p + 32);
            bk.q[1] = *(const uint4*)(k1p + 40);
            s1 = __builtin_amdgcn_wmma_f32_16x16x32_bf16(false, aq1.v, false, bk.v, (short)0, s1, false, false);
        }

        // ---- add edge bias + additive mask + key-padding mask
        const int key0 = j0 + l15, key1 = key0 + 16;
        float s0a[8], s1a[8];
#pragma unroll
        for (int r = 0; r < 8; ++r) {
            const int m = r + 8 * half;
            const int q = q0 + m;
            const int* er = edge + ((size_t)b * Lq + q) * Lq;
            float b0 = etab[er[key0] & 15] + attn_mask[q * Lq + key0];
            float b1 = etab[er[key1] & 15] + attn_mask[q * Lq + key1];
            if (kpm[b * Lq + key0]) b0 = NEG_INF;
            if (kpm[b * Lq + key1]) b1 = NEG_INF;
            s0a[r] = s0[r] + b0;
            s1a[r] = s1[r] + b1;
        }

        // ---- online softmax update (row lives in one 16-lane half)
#pragma unroll
        for (int r = 0; r < 8; ++r) {
            float mx = fmaxf(s0a[r], s1a[r]);
#pragma unroll
            for (int off = 1; off < 16; off <<= 1)
                mx = fmaxf(mx, __shfl_xor(mx, off, 32));
            const float mnew = fmaxf(mrow[r], mx);
            float alpha, p0, p1;
            if (mnew == NEG_INF) { alpha = 1.f; p0 = 0.f; p1 = 0.f; }
            else {
                alpha = __expf(mrow[r] - mnew);
                p0 = __expf(s0a[r] - mnew);
                p1 = __expf(s1a[r] - mnew);
            }
            float ps = p0 + p1;
#pragma unroll
            for (int off = 1; off < 16; off <<= 1)
                ps += __shfl_xor(ps, off, 32);
            lrow[r] = lrow[r] * alpha + ps;
            mrow[r] = mnew;
#pragma unroll
            for (int c = 0; c < 4; ++c) o[c][r] *= alpha;
            const int m = r + 8 * half;
            Pl[m][l15]      = p0;
            Pl[m][16 + l15] = p1;
        }
        __syncthreads();

        // ---- P tile: C-layout -> A-layout via LDS
        Frag16 pa;
#pragma unroll
        for (int v = 0; v < 8; ++v) {
            const int ka = half * 16 + 2 * v;
            pa.d[v] = pack2(Pl[l15][ka], Pl[l15][ka + 1]);
        }
        __syncthreads();

        // ---- O += P * V  (four 16-wide d chunks); V^T rows give b128 loads
#pragma unroll
        for (int c = 0; c < 4; ++c) {
            const unsigned short* vr =
                Vt + ((size_t)bh * Dh + c * 16 + l15) * Lq + j0 + half * 16;
            if (c == 0 && j0 + 32 < Lq) __builtin_prefetch(vr + 32, 0, 0);
            Frag16 vb;
            vb.q[0] = *(const uint4*)(vr);
            vb.q[1] = *(const uint4*)(vr + 8);
            o[c] = __builtin_amdgcn_wmma_f32_16x16x32_bf16(false, pa.v, false, vb.v,
                                                           (short)0, o[c], false, false);
        }
    }

    // ---- normalize and store as bf16 in [L*B, 512] layout for out-proj
#pragma unroll
    for (int r = 0; r < 8; ++r) {
        const int m = r + 8 * half;
        const int q = q0 + m;
        const float inv = (lrow[r] > 0.f) ? 1.0f / lrow[r] : 0.f;
        const size_t base = ((size_t)q * Bc + b) * Ec + h * Dh;
#pragma unroll
        for (int c = 0; c < 4; ++c)
            AO[base + c * 16 + l15] = f2bf(o[c][r] * inv);
    }
}

// ============================================================
// Kernel 3: out = AO(bf16)[8192,512] @ out_proj_w.T + b -> fp32
// ============================================================
__global__ __launch_bounds__(256)
void outproj_kernel(const unsigned short* __restrict__ AO,
                    const float* __restrict__ W,
                    const float* __restrict__ bias,
                    float* __restrict__ out)
{
    const int lane = threadIdx.x & 31;
    const int wid  = blockIdx.x * (blockDim.x >> 5) + (threadIdx.x >> 5);
    const int TN = Ec / 16;                    // 32 col tiles
    const int tile_m = wid / TN, tile_n = wid % TN;
    if (tile_m >= (Lq * Bc) / 16) return;
    const int m0 = tile_m * 16, n0 = tile_n * 16;
    const int half = lane >> 4, l15 = lane & 15;

    v8f acc = {};
    const unsigned short* arow = AO + (size_t)(m0 + l15) * Ec;
    const float* brow = W + (size_t)(n0 + l15) * Ec;
    for (int k0 = 0; k0 < Ec; k0 += 32) {
        Frag16 a, b;
        a.q[0] = *(const uint4*)(arow + k0 + half * 8);        // b128 A loads
        a.q[1] = *(const uint4*)(arow + k0 + 16 + half * 8);
#pragma unroll
        for (int v = 0; v < 8; ++v) {
            const int kb = k0 + ((v >> 2) * 16) + half * 8 + ((v & 3) * 2);
            b.d[v] = pack2(brow[kb], brow[kb + 1]);
        }
        acc = __builtin_amdgcn_wmma_f32_16x16x32_bf16(false, a.v, false, b.v,
                                                      (short)0, acc, false, false);
    }
#pragma unroll
    for (int r = 0; r < 8; ++r) {
        const int m = r + 8 * half;
        out[(size_t)(m0 + m) * Ec + n0 + l15] = acc[r] + bias[n0 + l15];
    }
}

// ============================================================
extern "C" void kernel_launch(void* const* d_in, const int* in_sizes, int n_in,
                              void* d_out, int out_size, void* d_ws, size_t ws_size,
                              hipStream_t stream)
{
    const float*         x      = (const float*)d_in[0];
    const int*           edge   = (const int*)d_in[1];          // jax default: int32
    const float*         amask  = (const float*)d_in[2];
    const unsigned char* kpm    = (const unsigned char*)d_in[3];
    const float*         ipw    = (const float*)d_in[4];        // [1536,512]
    const float*         ipb    = (const float*)d_in[5];
    const float*         opw    = (const float*)d_in[6];        // [512,512]
    const float*         opb    = (const float*)d_in[7];
    const float*         etab   = (const float*)d_in[8];        // [16,8]
    float* out = (float*)d_out;

    const size_t NELEM = (size_t)Bc * Hc * Lq * Dh;             // 4M bf16 each
    unsigned short* Qb = (unsigned short*)d_ws;
    unsigned short* Kb = Qb + NELEM;
    unsigned short* Vt = Kb + NELEM;
    unsigned short* AO = Vt + NELEM;

    // 1) QKV projection: 512 x 96 tiles, 8 waves / block
    {
        const int tiles = ((Lq * Bc) / 16) * ((3 * Ec) / 16);   // 49152
        qkv_kernel<<<tiles / 8, 256, 0, stream>>>(x, ipw, ipb, Qb, Kb, Vt);
    }
    // 2) fused attention: one wave per (b,h,qblock)
    {
        const int blocks = Bc * Hc * (Lq / 16);                 // 4096
        attn_kernel<<<blocks, 32, 0, stream>>>(Qb, Kb, Vt, edge, amask, kpm, etab, AO);
    }
    // 3) output projection: 512 x 32 tiles, 8 waves / block
    {
        const int tiles = ((Lq * Bc) / 16) * (Ec / 16);         // 16384
        outproj_kernel<<<tiles / 8, 256, 0, stream>>>(AO, opw, opb, out);
    }
}
